// SLAHead_10462540333852
// MI455X (gfx1250) — matrix operand order repaired and checked by reading, not compiled, not run on hardware
//
#include <hip/hip_runtime.h>

// ---------------------------------------------------------------------------
// SLAHead decoder for MI455X (gfx1250, wave32, WMMA bf16)
// ---------------------------------------------------------------------------
#define NB 64          // batch
#define NC 512         // channels
#define NT 1024        // spatial positions (32x32)
#define NH 256         // HID
#define NE 30          // NUM_EMB
#define NS 501         // decode steps
#define G3 768         // 3*HID

typedef __attribute__((ext_vector_type(16))) __bf16 v16bf;
typedef __attribute__((ext_vector_type(8)))  __bf16 v8bf;
typedef __attribute__((ext_vector_type(8)))  float  v8f;

// ----- WMMA fragment helpers (layouts per CDNA5 ISA 7.12.2, wave32) --------

// A: 16x32 bf16, row-major source with row stride lda (elements).
__device__ __forceinline__ v16bf load_A_frag(const __bf16* A, int lda) {
    int lane = threadIdx.x & 31;
    int m    = lane & 15;
    int kg   = (lane >> 4) * 8;
    const __bf16* row = A + (size_t)m * lda;
    v8bf lo = *(const v8bf*)(row + kg);
    v8bf hi = *(const v8bf*)(row + 16 + kg);
    v16bf f;
#pragma unroll
    for (int i = 0; i < 8; ++i) { f[i] = lo[i]; f[8 + i] = hi[i]; }
    return f;
}

// B: 32x16 bf16 from pre-transposed weights Bt stored [N][K] row-major.
__device__ __forceinline__ v16bf load_Bt_frag(const __bf16* Bt, int ldb) {
    int lane = threadIdx.x & 31;
    int n    = lane & 15;
    int kb   = (lane >> 4) * 8;
    const __bf16* row = Bt + (size_t)n * ldb + kb;
    v8bf lo = *(const v8bf*)(row);
    v8bf hi = *(const v8bf*)(row + 16);
    v16bf f;
#pragma unroll
    for (int i = 0; i < 8; ++i) { f[i] = lo[i]; f[8 + i] = hi[i]; }
    return f;
}

__device__ __forceinline__ void store_tile_f32(float* D, int ldd, v8f d) {
    int lane = threadIdx.x & 31;
    int n = lane & 15, mb = (lane >> 4) * 8;
#pragma unroll
    for (int i = 0; i < 8; ++i) D[(size_t)(mb + i) * ldd + n] = d[i];
}
__device__ __forceinline__ void store_tile_f32_bias(float* D, int ldd, v8f d,
                                                    const float* bias) {
    int lane = threadIdx.x & 31;
    int n = lane & 15, mb = (lane >> 4) * 8;
    float bv = bias[n];
#pragma unroll
    for (int i = 0; i < 8; ++i) D[(size_t)(mb + i) * ldd + n] = d[i] + bv;
}
__device__ __forceinline__ void store_tile_bf16(__bf16* D, int ldd, v8f d) {
    int lane = threadIdx.x & 31;
    int n = lane & 15, mb = (lane >> 4) * 8;
#pragma unroll
    for (int i = 0; i < 8; ++i) D[(size_t)(mb + i) * ldd + n] = (__bf16)d[i];
}

__device__ __forceinline__ v8f wmma_bf16(v16bf a, v16bf b, v8f c) {
    return __builtin_amdgcn_wmma_f32_16x16x32_bf16(false, a, false, b,
                                                   (short)0, c, false, false);
}

__device__ __forceinline__ float wave_sum(float v) {
#pragma unroll
    for (int off = 16; off > 0; off >>= 1) v += __shfl_xor(v, off, 32);
    return v;
}
__device__ __forceinline__ float wave_max(float v) {
#pragma unroll
    for (int off = 16; off > 0; off >>= 1) v = fmaxf(v, __shfl_xor(v, off, 32));
    return v;
}

// fast tanh/sigmoid on the v_exp_f32/v_rcp_f32 trans pipe (hot path: 16.7M/step)
__device__ __forceinline__ float fast_tanh(float x) {
    float xc = fminf(fmaxf(x, -9.f), 9.f);
    float e  = __expf(2.f * xc);
    return (e - 1.f) * __builtin_amdgcn_rcpf(e + 1.f);
}
__device__ __forceinline__ float sigm(float x) {
    return __builtin_amdgcn_rcpf(1.f + __expf(-x));
}

// ---------------------------------------------------------------------------
// Prep kernels
// ---------------------------------------------------------------------------

// src[rows][cols] f32 (row stride == cols) -> dst[cols][rows] bf16
__global__ void k_transpose_bf16(const float* __restrict__ src,
                                 __bf16* __restrict__ dst, int rows, int cols) {
    int i = blockIdx.x * blockDim.x + threadIdx.x;
    if (i < rows * cols) {
        int r = i / cols, c = i - r * cols;
        dst[(size_t)c * rows + r] = (__bf16)src[i];
    }
}

__global__ void k_copy_emb(const float* __restrict__ w_ih, float* __restrict__ embR) {
    int i = blockIdx.x * blockDim.x + threadIdx.x;          // NE*G3
    if (i < NE * G3) embR[i] = w_ih[(size_t)NC * G3 + i];   // rows 512..541
}

__global__ void k_convert_fea(const float* __restrict__ fea, __bf16* __restrict__ feaB) {
    size_t base = ((size_t)blockIdx.x * blockDim.x + threadIdx.x) * 8;
#pragma unroll
    for (int j = 0; j < 8; ++j) feaB[base + j] = (__bf16)fea[base + j];
}

// hidden=0; ph0 = h2h_b (since h0 == 0)
__global__ void k_init(float* hidF, __bf16* hidB, float* ph,
                       const float* __restrict__ h2h_b) {
    int i = blockIdx.x * blockDim.x + threadIdx.x;          // NB*NH
    hidF[i] = 0.f;
    hidB[i] = (__bf16)0.f;
    ph[i]   = h2h_b[i & (NH - 1)];
}

// proj[b][t][h] = sum_c fea[b][c][t] * i2h_w[c][h] ; A transposed through LDS.
__global__ void k_proj(const float* __restrict__ fea,
                       const __bf16* __restrict__ i2hT,   // [NH][NC]
                       __bf16* __restrict__ proj) {
    __shared__ __attribute__((aligned(16))) __bf16 As[64][48]; // [t][k], 96B rows
    int b  = blockIdx.x >> 4;
    int t0 = (blockIdx.x & 15) * 64;
    int wave = threadIdx.x >> 5;

    v8f acc[8];
#pragma unroll
    for (int i = 0; i < 8; ++i) acc[i] = (v8f){};

    for (int kc = 0; kc < NC; kc += 32) {
        __syncthreads();
        int cc = threadIdx.x >> 3;            // 0..31
        int tt = (threadIdx.x & 7) * 8;       // 0..56
        const float* srow = fea + ((size_t)b * NC + kc + cc) * NT + t0 + tt;
#pragma unroll
        for (int j = 0; j < 8; ++j) As[tt + j][cc] = (__bf16)srow[j];
        __syncthreads();
#pragma unroll
        for (int i = 0; i < 8; ++i) {
            int id = wave + 8 * i;            // 0..63
            int mt = id & 3, nt = id >> 2;    // 4 M-tiles x 16 N-tiles
            v16bf a = load_A_frag(&As[mt * 16][0], 48);
            v16bf bb = load_Bt_frag(i2hT + (size_t)nt * 16 * NC + kc, NC);
            acc[i] = wmma_bf16(a, bb, acc[i]);
        }
    }
#pragma unroll
    for (int i = 0; i < 8; ++i) {
        int id = wave + 8 * i;
        int mt = id & 3, nt = id >> 2;
        store_tile_bf16(proj + ((size_t)b * NT + t0 + mt * 16) * NH + nt * 16,
                        NH, acc[i]);
    }
}

// ---------------------------------------------------------------------------
// Per-step kernels
// ---------------------------------------------------------------------------

// e[b][t] = sum_h score_w[h] * tanh(proj[b][t][h] + ph[b][h])
// 8 rows (same b) per wave; score_w & ph cached in registers across rows.
__global__ void k_scores(const __bf16* __restrict__ proj,
                         const float* __restrict__ ph,
                         const float* __restrict__ score_w,
                         float* __restrict__ evec) {
    int gw = blockIdx.x * 8 + (threadIdx.x >> 5);   // 0..8191
    int lane = threadIdx.x & 31;
    int r0 = gw * 8;                                // first row; b fixed over 8
    int b = r0 >> 10;
    int h0 = lane * 8;

    float sw[8], phb[8];
#pragma unroll
    for (int j = 0; j < 8; ++j) {
        sw[j]  = score_w[h0 + j];
        phb[j] = ph[b * NH + h0 + j];
    }
    const __bf16* prow = proj + (size_t)r0 * NH + h0;
#pragma unroll
    for (int r = 0; r < 8; ++r) {
        v8bf p = *(const v8bf*)(prow + (size_t)r * NH);
        float acc = 0.f;
#pragma unroll
        for (int j = 0; j < 8; ++j) acc += sw[j] * fast_tanh((float)p[j] + phb[j]);
        acc = wave_sum(acc);
        if (lane == 0) evec[r0 + r] = acc;
    }
}

// per-b softmax over T, then context[b][c] = sum_t alpha_t * feat[b][c][t]
__global__ void k_softmax_ctx(const float* __restrict__ evec,
                              const __bf16* __restrict__ feaB,
                              __bf16* __restrict__ ctxB) {
    __shared__ float alpha[NT];
    __shared__ float red[24];
    int b = blockIdx.x, tid = threadIdx.x;      // 256 threads, 8 waves
    int wv = tid >> 5, ln = tid & 31;

    float v[4], m = -1e30f;
#pragma unroll
    for (int j = 0; j < 4; ++j) { v[j] = evec[b * NT + tid + j * 256]; m = fmaxf(m, v[j]); }
    m = wave_max(m);
    if (ln == 0) red[wv] = m;
    __syncthreads();
    if (tid == 0) { float mm = red[0]; for (int i = 1; i < 8; ++i) mm = fmaxf(mm, red[i]); red[16] = mm; }
    __syncthreads();
    m = red[16];

    float s = 0.f;
#pragma unroll
    for (int j = 0; j < 4; ++j) { float ex = __expf(v[j] - m); alpha[tid + j * 256] = ex; s += ex; }
    s = wave_sum(s);
    if (ln == 0) red[8 + wv] = s;
    __syncthreads();
    if (tid == 0) { float ss = 0.f; for (int i = 0; i < 8; ++i) ss += red[8 + i]; red[17] = 1.f / ss; }
    __syncthreads();
    float inv = red[17];
#pragma unroll
    for (int j = 0; j < 4; ++j) alpha[tid + j * 256] *= inv;
    __syncthreads();

    // hoist this lane's 32 alpha values (loop-invariant over channels) to regs
    float av[32];
#pragma unroll
    for (int ch = 0; ch < 4; ++ch)
#pragma unroll
        for (int j = 0; j < 8; ++j) av[ch * 8 + j] = alpha[ch * 256 + ln * 8 + j];

    // context: wave handles 64 channels, lanes split T
    int c0 = wv * 64;
    for (int i = 0; i < 64; ++i) {
        const __bf16* fr = feaB + ((size_t)b * NC + c0 + i) * NT;
        if (i + 1 < 64) __builtin_prefetch(fr + NT, 0, 1);  // global_prefetch_b8
        float acc = 0.f;
#pragma unroll
        for (int ch = 0; ch < 4; ++ch) {
            int t = ch * 256 + ln * 8;
            v8bf f = *(const v8bf*)(fr + t);
#pragma unroll
            for (int j = 0; j < 8; ++j) acc += (float)f[j] * av[ch * 8 + j];
        }
        acc = wave_sum(acc);
        if (ln == 0) ctxB[b * NC + c0 + i] = (__bf16)acc;
    }
}

// GRU cell + ph_{s+1} + head trunks + output heads. Single 1024-thread block.
__global__ void k_step(const int* __restrict__ targets,
                       const __bf16* __restrict__ ctxB,
                       float* __restrict__ hidF, __bf16* __restrict__ hidB,
                       const __bf16* __restrict__ wihT,  // [G3][NC]
                       const __bf16* __restrict__ whhT,  // [G3][NH]
                       const __bf16* __restrict__ h2hT,  // [NH][NH]
                       const __bf16* __restrict__ sg1T,
                       const __bf16* __restrict__ lg1T,
                       const float* __restrict__ embR,   // [NE][G3]
                       const float* __restrict__ b_ih, const float* __restrict__ b_hh,
                       const float* __restrict__ h2h_b,
                       const float* __restrict__ sg1_b, const float* __restrict__ lg1_b,
                       const float* __restrict__ sg2_w, const float* __restrict__ sg2_b,
                       const float* __restrict__ lg2_w, const float* __restrict__ lg2_b,
                       float* __restrict__ gx, float* __restrict__ gh,
                       float* __restrict__ ph, float* __restrict__ h1w,
                       float* __restrict__ l1w, float* __restrict__ out, int s) {
    int wave = threadIdx.x >> 5;

    // --- C1: gx = ctx @ W_ih[:512]  (4x48 tiles, K=512)
    //         gh = h   @ W_hh        (4x48 tiles, K=256)
    for (int ti = wave; ti < 384; ti += 32) {
        bool isGx = ti < 192;
        int id = isGx ? ti : ti - 192;
        int mt = id & 3, nt = id >> 2;
        const __bf16* A  = isGx ? (ctxB + mt * 16 * NC) : ((const __bf16*)hidB + mt * 16 * NH);
        const __bf16* Bt = isGx ? (wihT + (size_t)nt * 16 * NC) : (whhT + (size_t)nt * 16 * NH);
        int K = isGx ? NC : NH;
        v8f acc = (v8f){};
        for (int k = 0; k < K; k += 32) {
            v16bf a  = load_A_frag(A + k, K);
            v16bf bb = load_Bt_frag(Bt + k, K);
            acc = wmma_bf16(a, bb, acc);
        }
        store_tile_f32((isGx ? gx : gh) + (size_t)(mt * 16) * G3 + nt * 16, G3, acc);
    }
    __threadfence();
    __syncthreads();

    // --- C2: gate math, h_new (one-hot == row select from gru_w_ih[512+char])
    for (int idx = threadIdx.x; idx < NB * NH; idx += 1024) {
        int b = idx >> 8, n = idx & (NH - 1);
        int ch = targets[b * NS + s];
        const float* em = embR + (size_t)ch * G3;
        const float* gxb = gx + (size_t)b * G3;
        const float* ghb = gh + (size_t)b * G3;
        float xr = gxb[n]        + b_ih[n]        + em[n];
        float xz = gxb[256 + n]  + b_ih[256 + n]  + em[256 + n];
        float xn = gxb[512 + n]  + b_ih[512 + n]  + em[512 + n];
        float hr = ghb[n]        + b_hh[n];
        float hz = ghb[256 + n]  + b_hh[256 + n];
        float hn = ghb[512 + n]  + b_hh[512 + n];
        float r = sigm(xr + hr);
        float z = sigm(xz + hz);
        float nn = fast_tanh(xn + r * hn);
        float hnew = (1.f - z) * nn + z * hidF[idx];
        hidF[idx] = hnew;
        hidB[idx] = (__bf16)hnew;
    }
    __threadfence();
    __syncthreads();

    // --- C3: ph_{s+1} = h@h2h + b ; h1 = h@sg1 + b ; l1 = h@lg1 + b
    for (int ti = wave; ti < 192; ti += 32) {
        int which = ti >> 6;              // 0:ph 1:h1 2:l1
        int id = ti & 63;
        int mt = id & 3, nt = id >> 2;
        const __bf16* Bt = (which == 0) ? h2hT : (which == 1) ? sg1T : lg1T;
        const float* bias = (which == 0) ? h2h_b : (which == 1) ? sg1_b : lg1_b;
        float* D = (which == 0) ? ph : (which == 1) ? h1w : l1w;
        v8f acc = (v8f){};
        for (int k = 0; k < NH; k += 32) {
            v16bf a  = load_A_frag((const __bf16*)hidB + mt * 16 * NH + k, NH);
            v16bf bb = load_Bt_frag(Bt + (size_t)nt * 16 * NH + k, NH);
            acc = wmma_bf16(a, bb, acc);
        }
        store_tile_f32_bias(D + (size_t)(mt * 16) * NH + nt * 16, NH, acc,
                            bias + nt * 16);
    }
    __threadfence();
    __syncthreads();

    // --- C4: heads (tiny N: scalar dots), write d_out at step s
    const size_t LOC_BASE = (size_t)NB * NS * NE;
    for (int w = threadIdx.x; w < NB * NE + NB * 4; w += 1024) {
        if (w < NB * NE) {
            int b = w / NE, k = w - b * NE;
            const float* hb = h1w + (size_t)b * NH;
            float acc = sg2_b[k];
            for (int h = 0; h < NH; ++h) acc += hb[h] * sg2_w[h * NE + k];
            out[((size_t)b * NS + s) * NE + k] = acc;
        } else {
            int j = w - NB * NE;
            int b = j >> 2, q = j & 3;
            const float* lb = l1w + (size_t)b * NH;
            float acc = lg2_b[q];
            for (int h = 0; h < NH; ++h) acc += lb[h] * lg2_w[h * 4 + q];
            out[LOC_BASE + ((size_t)b * NS + s) * 4 + q] = sigm(acc);
        }
    }
}

// ---------------------------------------------------------------------------
extern "C" void kernel_launch(void* const* d_in, const int* in_sizes, int n_in,
                              void* d_out, int out_size, void* d_ws, size_t ws_size,
                              hipStream_t stream) {
    const float* fea      = (const float*)d_in[0];
    const int*   targets  = (const int*)d_in[1];
    const float* i2h_w    = (const float*)d_in[2];
    const float* h2h_w    = (const float*)d_in[3];
    const float* h2h_b    = (const float*)d_in[4];
    const float* score_w  = (const float*)d_in[5];
    const float* gru_w_ih = (const float*)d_in[6];
    const float* gru_w_hh = (const float*)d_in[7];
    const float* gru_b_ih = (const float*)d_in[8];
    const float* gru_b_hh = (const float*)d_in[9];
    const float* sg1_w    = (const float*)d_in[10];
    const float* sg1_b    = (const float*)d_in[11];
    const float* sg2_w    = (const float*)d_in[12];
    const float* sg2_b    = (const float*)d_in[13];
    const float* lg1_w    = (const float*)d_in[14];
    const float* lg1_b    = (const float*)d_in[15];
    const float* lg2_w    = (const float*)d_in[16];
    const float* lg2_b    = (const float*)d_in[17];
    float* out = (float*)d_out;

    char* wp = (char*)d_ws;
    auto take = [&](size_t bytes) {
        char* p = wp; wp += (bytes + 255) & ~(size_t)255; return p;
    };
    __bf16* feaB = (__bf16*)take((size_t)NB * NC * NT * 2);   // 67 MB (L2-resident)
    __bf16* proj = (__bf16*)take((size_t)NB * NT * NH * 2);   // 33.5 MB (L2-resident)
    __bf16* i2hT = (__bf16*)take((size_t)NH * NC * 2);
    __bf16* h2hT = (__bf16*)take((size_t)NH * NH * 2);
    __bf16* wihT = (__bf16*)take((size_t)G3 * NC * 2);
    __bf16* whhT = (__bf16*)take((size_t)G3 * NH * 2);
    __bf16* sg1T = (__bf16*)take((size_t)NH * NH * 2);
    __bf16* lg1T = (__bf16*)take((size_t)NH * NH * 2);
    float*  embR = (float*)take((size_t)NE * G3 * 4);
    float*  hidF = (float*)take((size_t)NB * NH * 4);
    __bf16* hidB = (__bf16*)take((size_t)NB * NH * 2);
    float*  ph   = (float*)take((size_t)NB * NH * 4);
    float*  evec = (float*)take((size_t)NB * NT * 4);
    __bf16* ctxB = (__bf16*)take((size_t)NB * NC * 2);
    float*  gx   = (float*)take((size_t)NB * G3 * 4);
    float*  gh   = (float*)take((size_t)NB * G3 * 4);
    float*  h1w  = (float*)take((size_t)NB * NH * 4);
    float*  l1w  = (float*)take((size_t)NB * NH * 4);

    // ---- one-time prep: bf16 weight transposes, feat convert, proj GEMM ----
    auto tr = [&](const float* src, __bf16* dst, int rows, int cols) {
        int total = rows * cols;
        k_transpose_bf16<<<(total + 255) / 256, 256, 0, stream>>>(src, dst, rows, cols);
    };
    tr(i2h_w, i2hT, NC, NH);          // -> [NH][NC]
    tr(h2h_w, h2hT, NH, NH);
    tr(gru_w_ih, wihT, NC, G3);       // first 512 rows -> [G3][NC]
    tr(gru_w_hh, whhT, NH, G3);       // -> [G3][NH]
    tr(sg1_w, sg1T, NH, NH);
    tr(lg1_w, lg1T, NH, NH);
    k_copy_emb<<<(NE * G3 + 255) / 256, 256, 0, stream>>>(gru_w_ih, embR);
    {
        size_t total = (size_t)NB * NC * NT / 8;
        k_convert_fea<<<(unsigned)(total / 256), 256, 0, stream>>>(fea, feaB);
    }
    k_init<<<NB * NH / 256, 256, 0, stream>>>(hidF, hidB, ph, h2h_b);
    k_proj<<<NB * 16, 256, 0, stream>>>(fea, i2hT, proj);

    // ---- 501 sequential decode steps --------------------------------------
    for (int s = 0; s < NS; ++s) {
        k_scores<<<(NB * NT) / 64, 256, 0, stream>>>(proj, ph, score_w, evec);
        k_softmax_ctx<<<NB, 256, 0, stream>>>(evec, feaB, ctxB);
        k_step<<<1, 1024, 0, stream>>>(targets, ctxB, hidF, hidB,
                                       wihT, whhT, h2hT, sg1T, lg1T, embR,
                                       gru_b_ih, gru_b_hh, h2h_b, sg1_b, lg1_b,
                                       sg2_w, sg2_b, lg2_w, lg2_b,
                                       gx, gh, ph, h1w, l1w, out, s);
    }
}
